// SurvModel_36326833389991
// MI455X (gfx1250) — compile-verified, matrix operand order checked by reading, not run on hardware
//
#include <hip/hip_runtime.h>
#include <hip/hip_bf16.h>
#include <math.h>

typedef __attribute__((ext_vector_type(16))) __bf16 v16bf;
typedef __attribute__((ext_vector_type(8)))  __bf16 v8bf;
typedef __attribute__((ext_vector_type(8)))  float  v8f;

#define N_ROWS 131072
#define DIN 768
#define DH 128
#define DA 32
#define TROWS 128
#define LDA 40     // padded bf16 stride for 32-wide K-chunk rows (80B, 16B aligned, bank-spread)
#define LDX 136    // padded bf16 stride for 128-wide x rows (272B)
#define BN_INV 0.9999950000374997f  // 1/sqrt(1+1e-5)

#ifndef USE_ASYNC_B
#define USE_ASYNC_B 1   // set to 0 if global_load_async_to_lds_* fails to assemble
#endif

static __device__ inline v16bf comb(v8bf a, v8bf b) {
    v16bf r;
#pragma unroll
    for (int i = 0; i < 8; ++i) { r[i] = a[i]; r[i + 8] = b[i]; }
    return r;
}

// ---------------------------------------------------------------- K0: weight prep
__global__ __launch_bounds__(256)
void prep_weights(const float* __restrict__ W1, const float* __restrict__ Watt1,
                  __bf16* __restrict__ W1T, __bf16* __restrict__ Wa1T) {
    int i = blockIdx.x * blockDim.x + threadIdx.x;
    if (i < DIN * DH) {                 // W1T[n][k] = W1[k][n]
        int n = i / DIN, k = i % DIN;
        W1T[i] = (__bf16)W1[k * DH + n];
    }
    int j = i - DIN * DH;
    if (j >= 0 && j < DA * DH) {        // Wa1T[a][n] = Watt1[n][a]
        int a = j / DH, n = j % DH;
        Wa1T[j] = (__bf16)Watt1[n * DA + a];
    }
}

// ---------------------------------------------------------------- K1: fused big GEMM + BN/ReLU + attn-score
__global__ __launch_bounds__(256)
void main_fused(const float* __restrict__ input,
                const __bf16* __restrict__ W1T,
                const __bf16* __restrict__ Wa1T,
                const float* __restrict__ b1, const float* __restrict__ g1, const float* __restrict__ be1,
                const float* __restrict__ batt1, const float* __restrict__ g2, const float* __restrict__ be2,
                const float* __restrict__ Watt2, const float* __restrict__ batt2,
                float* __restrict__ scores) {
    union SharedU {
        struct { __bf16 A[2][TROWS * LDA]; __bf16 B[2][DH * LDA]; } ab;   // 40960 B
        __bf16 X[8][16 * LDX];                                            // 34816 B
    };
    __shared__ SharedU sh;

    const int tid   = threadIdx.x;
    const int lane  = tid & 31;
    const int wave  = tid >> 5;
    const int lrow  = lane & 15;
    const int lhalf = lane >> 4;
    const int rowBase = blockIdx.x * TROWS;

    // global->LDS chunk loader assignment: thread covers 16 contiguous elements of one row
    const int lr = tid >> 1;              // row 0..127
    const int lo = (tid & 1) * 16;        // K offset 0 or 16 within chunk
    const float*  aptr = input + (size_t)(rowBase + lr) * DIN + lo;
    const __bf16* bptr = W1T + (size_t)lr * DIN + lo;

    v8f acc[8];
#pragma unroll
    for (int i = 0; i < 8; ++i)
#pragma unroll
        for (int j = 0; j < 8; ++j) acc[i][j] = 0.f;

    auto cvt8 = [](float4 a, float4 b) -> v8bf {
        v8bf r;
        r[0] = (__bf16)a.x; r[1] = (__bf16)a.y; r[2] = (__bf16)a.z; r[3] = (__bf16)a.w;
        r[4] = (__bf16)b.x; r[5] = (__bf16)b.y; r[6] = (__bf16)b.z; r[7] = (__bf16)b.w;
        return r;
    };

    // ---- split fetch (issue loads, no wait) from commit (wait + convert + LDS store)
    float4 fa0, fa1, fa2, fa3;
#if !USE_ASYNC_B
    v8bf qb0, qb1;
#endif
    auto fetchA = [&](int kc) {
        const float* ap = aptr + kc * 32;
        fa0 = *(const float4*)(ap + 0);
        fa1 = *(const float4*)(ap + 4);
        fa2 = *(const float4*)(ap + 8);
        fa3 = *(const float4*)(ap + 12);
    };
    auto commitA = [&](int buf) {
        *(v8bf*)&sh.ab.A[buf][lr * LDA + lo]     = cvt8(fa0, fa1);
        *(v8bf*)&sh.ab.A[buf][lr * LDA + lo + 8] = cvt8(fa2, fa3);
    };
#if USE_ASYNC_B
    // bf16 B chunk goes memory->LDS directly via the CDNA5 async data path (ASYNCcnt).
    // LDS destination operand = wave-relative LDS byte offset = low 32 bits of generic ptr.
    auto loadB_async = [&](int kc, int buf) {
        unsigned long long gp = (unsigned long long)(bptr + kc * 32);
        unsigned l0 = (unsigned)(size_t)(const void*)&sh.ab.B[buf][lr * LDA + lo];
        asm volatile("global_load_async_to_lds_b128 %0, %1, off"
                     :: "v"(l0), "v"(gp) : "memory");
        asm volatile("global_load_async_to_lds_b128 %0, %1, off"
                     :: "v"(l0 + 16u), "v"(gp + 16ull) : "memory");
    };
#else
    auto fetchB = [&](int kc) {
        const __bf16* bp = bptr + kc * 32;
        qb0 = *(const v8bf*)(bp);
        qb1 = *(const v8bf*)(bp + 8);
    };
    auto commitB = [&](int buf) {
        *(v8bf*)&sh.ab.B[buf][lr * LDA + lo]     = qb0;
        *(v8bf*)&sh.ab.B[buf][lr * LDA + lo + 8] = qb1;
    };
#endif

    // prologue: chunk 0
    fetchA(0);
#if USE_ASYNC_B
    loadB_async(0, 0);
#else
    fetchB(0);
#endif
    commitA(0);
#if !USE_ASYNC_B
    commitB(0);
#endif
#if USE_ASYNC_B
    asm volatile("s_wait_asynccnt 0x0" ::: "memory");
#endif
    __syncthreads();

    const int aoff = (wave * 16 + lrow) * LDA + lhalf * 8;

    for (int kc = 0; kc < 24; ++kc) {
        const int buf = kc & 1;
        const bool more = (kc + 1 < 24);
        if (more) {
            fetchA(kc + 1);          // issue global loads only (no wait yet)
#if USE_ASYNC_B
            loadB_async(kc + 1, buf ^ 1);
#else
            fetchB(kc + 1);
#endif
        }
        // batch ALL fragment LDS loads so one dscnt wait covers the group,
        // then the 8 independent WMMAs issue back-to-back
        v8bf a0 = *(const v8bf*)&sh.ab.A[buf][aoff];
        v8bf a1 = *(const v8bf*)&sh.ab.A[buf][aoff + 16];
        v8bf bl0[8], bl1[8];
#pragma unroll
        for (int nt = 0; nt < 8; ++nt) {
            const int boff = (nt * 16 + lrow) * LDA + lhalf * 8;
            bl0[nt] = *(const v8bf*)&sh.ab.B[buf][boff];
            bl1[nt] = *(const v8bf*)&sh.ab.B[buf][boff + 16];
        }
        v16bf af = comb(a0, a1);
#pragma unroll
        for (int nt = 0; nt < 8; ++nt) {
            acc[nt] = __builtin_amdgcn_wmma_f32_16x16x32_bf16(
                false, af, false, comb(bl0[nt], bl1[nt]), (short)0, acc[nt], false, false);
        }
        if (more) {
            commitA(buf ^ 1);        // loadcnt wait lands here, AFTER the WMMAs
#if !USE_ASYNC_B
            commitB(buf ^ 1);
#endif
        }
#if USE_ASYNC_B
        asm volatile("s_wait_asynccnt 0x0" ::: "memory");
#endif
        __syncthreads();
    }

    // ---- epilogue: BN + ReLU, stash x (bf16) into LDS (A/B buffers are dead now)
    __bf16* xw = &sh.X[wave][0];
#pragma unroll
    for (int nt = 0; nt < 8; ++nt) {
        const int n = nt * 16 + lrow;
        const float sc = g1[n] * BN_INV;
        const float bb = b1[n];
        const float bo = be1[n];
#pragma unroll
        for (int r = 0; r < 8; ++r) {
            const int m = r + 8 * lhalf;
            float v = (acc[nt][r] + bb) * sc + bo;
            v = v > 0.f ? v : 0.f;
            xw[m * LDX + n] = (__bf16)v;
        }
    }
    __syncthreads();

    // ---- h = tanh(bn(x @ Watt1)) via WMMA (A = x[16,128], B = Wa1T, 2 N-tiles, 4 K-steps)
    v8f hacc[2];
#pragma unroll
    for (int t2 = 0; t2 < 2; ++t2)
#pragma unroll
        for (int j = 0; j < 8; ++j) hacc[t2][j] = 0.f;
#pragma unroll
    for (int kc = 0; kc < 4; ++kc) {
        const int xo = lrow * LDX + kc * 32 + lhalf * 8;
        v8bf a0 = *(const v8bf*)&xw[xo];
        v8bf a1 = *(const v8bf*)&xw[xo + 16];
        v16bf af = comb(a0, a1);
#pragma unroll
        for (int t2 = 0; t2 < 2; ++t2) {
            const __bf16* wb = Wa1T + (t2 * 16 + lrow) * DH + kc * 32 + lhalf * 8;
            v8bf b0  = *(const v8bf*)(wb);
            v8bf b1v = *(const v8bf*)(wb + 16);
            v16bf bfr = comb(b0, b1v);
            hacc[t2] = __builtin_amdgcn_wmma_f32_16x16x32_bf16(
                false, af, false, bfr, (short)0, hacc[t2], false, false);
        }
    }

    // ---- s = h @ Watt2 (+batt2): per-lane partials, shfl-reduce across the 16-lane half
    float srow[8];
#pragma unroll
    for (int r = 0; r < 8; ++r) srow[r] = 0.f;
#pragma unroll
    for (int t2 = 0; t2 < 2; ++t2) {
        const int a = t2 * 16 + lrow;
        const float ba = batt1[a], sc2 = g2[a] * BN_INV, bo2 = be2[a], w2 = Watt2[a];
#pragma unroll
        for (int r = 0; r < 8; ++r) {
            float h = tanhf((hacc[t2][r] + ba) * sc2 + bo2);
            srow[r] += h * w2;
        }
    }
#pragma unroll
    for (int r = 0; r < 8; ++r) {
#pragma unroll
        for (int m = 1; m < 16; m <<= 1) srow[r] += __shfl_xor(srow[r], m, 32);
    }
    if (lrow == 0) {
        const float b2 = batt2[0];
#pragma unroll
        for (int r = 0; r < 8; ++r)
            scores[rowBase + wave * 16 + lhalf * 8 + r] = srow[r] + b2;
    }
}

// ---------------------------------------------------------------- K2: global softmax stats
__global__ __launch_bounds__(1024)
void softmax_stats(const float* __restrict__ scores, float* __restrict__ stats) {
    __shared__ float red[1024];
    const int t = threadIdx.x;
    float m = -1e30f;
    for (int i = t; i < N_ROWS; i += 1024) m = fmaxf(m, scores[i]);
    red[t] = m; __syncthreads();
    for (int s = 512; s > 0; s >>= 1) { if (t < s) red[t] = fmaxf(red[t], red[t + s]); __syncthreads(); }
    const float M = red[0]; __syncthreads();
    float sum = 0.f;
    for (int i = t; i < N_ROWS; i += 1024) sum += expf(scores[i] - M);
    red[t] = sum; __syncthreads();
    for (int s = 512; s > 0; s >>= 1) { if (t < s) red[t] += red[t + s]; __syncthreads(); }
    if (t == 0) { stats[0] = M; stats[1] = red[0]; }
}

// ---------------------------------------------------------------- K3: per-chunk top-64 (2048 -> 64)
__global__ __launch_bounds__(256)
void topk_stage1(const float* __restrict__ scores, float* __restrict__ candV, int* __restrict__ candI) {
    __shared__ float v[2048];
    __shared__ float rv[256];
    __shared__ int   ri[256];
    const int t = threadIdx.x;
    const int base = blockIdx.x * 2048;
    for (int i = t; i < 2048; i += 256) v[i] = scores[base + i];
    __syncthreads();
    for (int it = 0; it < 64; ++it) {
        float best = -1e30f; int bs = 0x7fffffff;
        for (int i = t; i < 2048; i += 256) {
            float x = v[i];
            if (x > best || (x == best && i < bs)) { best = x; bs = i; }
        }
        rv[t] = best; ri[t] = bs; __syncthreads();
        for (int s = 128; s > 0; s >>= 1) {
            if (t < s) {
                float xo = rv[t + s]; int io = ri[t + s];
                if (xo > rv[t] || (xo == rv[t] && io < ri[t])) { rv[t] = xo; ri[t] = io; }
            }
            __syncthreads();
        }
        if (t == 0) {
            candV[blockIdx.x * 64 + it] = rv[0];
            candI[blockIdx.x * 64 + it] = base + ri[0];
            v[ri[0]] = -1e30f;
        }
        __syncthreads();
    }
}

// ---------------------------------------------------------------- K4: candidates 4096 -> sorted top-64
__global__ __launch_bounds__(256)
void topk_stage2(const float* __restrict__ candV, const int* __restrict__ candI,
                 float* __restrict__ topV, int* __restrict__ topI) {
    __shared__ float v[4096];
    __shared__ int   gi[4096];
    __shared__ float rv[256];
    __shared__ int   ri[256];
    const int t = threadIdx.x;
    for (int i = t; i < 4096; i += 256) { v[i] = candV[i]; gi[i] = candI[i]; }
    __syncthreads();
    for (int it = 0; it < 64; ++it) {
        float best = -1e30f; int bs = 0x7fffffff;
        for (int i = t; i < 4096; i += 256) {
            float x = v[i];
            if (x > best || (x == best && i < bs)) { best = x; bs = i; }
        }
        rv[t] = best; ri[t] = bs; __syncthreads();
        for (int s = 128; s > 0; s >>= 1) {
            if (t < s) {
                float xo = rv[t + s]; int io = ri[t + s];
                if (xo > rv[t] || (xo == rv[t] && io < ri[t])) { rv[t] = xo; ri[t] = io; }
            }
            __syncthreads();
        }
        if (t == 0) {
            topV[it] = rv[0];
            topI[it] = gi[ri[0]];
            v[ri[0]] = -1e30f;
        }
        __syncthreads();
    }
}

// ---------------------------------------------------------------- K5: recompute x rows for top-64, apply double softmax weight
__global__ __launch_bounds__(128)
void compute_zs(const float* __restrict__ input, const float* __restrict__ W1,
                const float* __restrict__ b1, const float* __restrict__ g1, const float* __restrict__ be1,
                const float* __restrict__ topV, const int* __restrict__ topI,
                const float* __restrict__ stats, float* __restrict__ zs) {
    __shared__ float irow[DIN];
    __shared__ float Av[64];
    const int t = threadIdx.x;
    const int j = blockIdx.x;
    const float M = stats[0], S = stats[1];
    if (t < 64) Av[t] = expf(topV[t] - M) / S;   // A probabilities of the top-64 (sorted desc)
    const int row = topI[j];
    for (int i = t; i < DIN; i += 128) irow[i] = input[(size_t)row * DIN + i];
    __syncthreads();
    const float A0 = Av[0];
    float sum = 0.f;
    for (int i = 0; i < 64; ++i) sum += expf(Av[i] - A0);
    const float w = expf(Av[j] - A0) / sum;      // second softmax over probabilities
    float d = 0.f;
    for (int k = 0; k < DIN; ++k) d += irow[k] * W1[k * DH + t];
    float x = (d + b1[t]) * (g1[t] * BN_INV) + be1[t];
    x = x > 0.f ? x : 0.f;
    zs[j * DH + t] = x * w;
}

// ---------------------------------------------------------------- K6: tiny attention tail -> scalar output
__global__ __launch_bounds__(256)
void tail(const float* __restrict__ zs,
          const float* __restrict__ Wq, const float* __restrict__ Wk, const float* __restrict__ Wv,
          const float* __restrict__ Wz, const float* __restrict__ bz,
          const float* __restrict__ Wf, const float* __restrict__ bf,
          float* __restrict__ out) {
    __shared__ float zql[64 * 32], zkl[64 * 32], zvl[64 * 32], zwl[64 * 32];
    __shared__ float zl[128];
    __shared__ float red[256];
    const int t = threadIdx.x;
    for (int o = t; o < 2048; o += 256) {
        const int j = o >> 5, a = o & 31;
        float q = 0.f, k2 = 0.f, v2 = 0.f;
        const float* zr = zs + j * DH;
        for (int n = 0; n < DH; ++n) {
            float x = zr[n];
            q  += x * Wq[n * DA + a];
            k2 += x * Wk[n * DA + a];
            v2 += x * Wv[n * DA + a];
        }
        zql[o] = q; zkl[o] = k2; zvl[o] = v2;
    }
    __syncthreads();
    if (t < 64) {
        float lg[64]; float mx = -1e30f;
        for (int c = 0; c < 64; ++c) {
            float s = 0.f;
            for (int a = 0; a < DA; ++a) s += zql[t * DA + a] * zkl[c * DA + a];
            lg[c] = s; mx = fmaxf(mx, s);
        }
        float sm = 0.f;
        for (int c = 0; c < 64; ++c) { lg[c] = expf(lg[c] - mx); sm += lg[c]; }
        const float isn = 1.f / sm;
        for (int a = 0; a < DA; ++a) {
            float acc = 0.f;
            for (int c = 0; c < 64; ++c) acc += lg[c] * zvl[c * DA + a];
            zwl[t * DA + a] = acc * isn;
        }
    }
    __syncthreads();
    if (t < 128) {
        float acc = 0.f;
        for (int q = 0; q < 2048; ++q) acc += zwl[q] * Wz[q * DH + t];
        acc += bz[t];
        zl[t] = acc > 0.f ? acc : 0.f;
    }
    __syncthreads();
    float p = (t < 128) ? zl[t] * Wf[t] : 0.f;
    red[t] = p; __syncthreads();
    for (int s = 128; s > 0; s >>= 1) { if (t < s) red[t] += red[t + s]; __syncthreads(); }
    if (t == 0) out[0] = red[0] + bf[0];
}

// ---------------------------------------------------------------- launch
extern "C" void kernel_launch(void* const* d_in, const int* in_sizes, int n_in,
                              void* d_out, int out_size, void* d_ws, size_t ws_size,
                              hipStream_t stream) {
    const float* input = (const float*)d_in[0];
    const float* W1    = (const float*)d_in[1];
    const float* b1    = (const float*)d_in[2];
    const float* g1    = (const float*)d_in[3];
    const float* be1   = (const float*)d_in[4];
    const float* Watt1 = (const float*)d_in[5];
    const float* batt1 = (const float*)d_in[6];
    const float* g2    = (const float*)d_in[7];
    const float* be2   = (const float*)d_in[8];
    const float* Watt2 = (const float*)d_in[9];
    const float* batt2 = (const float*)d_in[10];
    const float* Wq    = (const float*)d_in[11];
    const float* Wk    = (const float*)d_in[12];
    const float* Wv    = (const float*)d_in[13];
    const float* Wz    = (const float*)d_in[14];
    const float* bz    = (const float*)d_in[15];
    const float* Wf    = (const float*)d_in[16];
    const float* bf    = (const float*)d_in[17];

    char* ws = (char*)d_ws;
    __bf16* W1T    = (__bf16*)(ws + 0);        // 196608 B
    __bf16* Wa1T   = (__bf16*)(ws + 196608);   //   8192 B
    float*  scores = (float*)(ws + 204800);    // 524288 B
    float*  stats  = (float*)(ws + 729088);    //    256 B
    float*  candV  = (float*)(ws + 729344);    //  16384 B
    int*    candI  = (int*)  (ws + 745728);    //  16384 B
    float*  topV   = (float*)(ws + 762112);    //    256 B
    int*    topI   = (int*)  (ws + 762368);    //    256 B
    float*  zs     = (float*)(ws + 762624);    //  32768 B

    prep_weights<<<400, 256, 0, stream>>>(W1, Watt1, W1T, Wa1T);
    main_fused<<<N_ROWS / TROWS, 256, 0, stream>>>(input, W1T, Wa1T,
                                                   b1, g1, be1, batt1, g2, be2,
                                                   Watt2, batt2, scores);
    softmax_stats<<<1, 1024, 0, stream>>>(scores, stats);
    topk_stage1<<<64, 256, 0, stream>>>(scores, candV, candI);
    topk_stage2<<<1, 256, 0, stream>>>(candV, candI, topV, topI);
    compute_zs<<<64, 128, 0, stream>>>(input, W1, b1, g1, be1, topV, topI, stats, zs);
    tail<<<1, 256, 0, stream>>>(zs, Wq, Wk, Wv, Wz, bz, Wf, bf, (float*)d_out);
}
